// MonTi_12902081757596
// MI455X (gfx1250) — matrix operand order repaired and checked by reading, not compiled, not run on hardware
//
#include <hip/hip_runtime.h>
#include <hip/hip_bf16.h>

#define NN        100000
#define TT        16
#define FEAT      512
#define EMB       256
#define FFD       1024
#define LAY       6
#define NH        4
#define DH        64
#define MAXC      128
#define ETGTN     262144
#define NBD       5
#define EBD       50
#define SEQ       149          // 16 + 128 + 5
#define SEQP      160          // padded to 16
#define START_ATK 144
#define OUTW      (FEAT + SEQ) // 661

typedef __attribute__((ext_vector_type(16))) _Float16 v16h;
typedef __attribute__((ext_vector_type(8)))  _Float16 v8h;
typedef __attribute__((ext_vector_type(8)))  float    v8f;

// ---------------- WMMA fragment loaders (wave32, contiguous k order in A and B) ----------------
// 16-bit A 16x32: lanes 0-15 row M=lane hold K=0..15, lanes 16-31 same rows
// hold K=16..31.  16B-aligned vector loads from LDS.
__device__ __forceinline__ v16h ld_a_lds(const _Float16* A, int ldA, int k0, int lane) {
  int row = lane & 15;
  int half = (lane >> 4) & 1;
  const _Float16* p = A + row * ldA + k0 + half * 16;
  v8h lo = *(const v8h*)p;
  v8h hi = *(const v8h*)(p + 8);
  v16h a;
#pragma unroll
  for (int i = 0; i < 8; ++i) { a[i] = lo[i]; a[8 + i] = hi[i]; }
  return a;
}

__device__ __forceinline__ v16h ld_a_g32(const float* A, int ldA, int row0, int k0, int lane) {
  int row = row0 + (lane & 15);
  int half = (lane >> 4) & 1;
  const float* p = A + (long)row * ldA + k0 + half * 16;
  v16h a;
#pragma unroll
  for (int i = 0; i < 8; ++i) a[i] = (_Float16)p[i];
#pragma unroll
  for (int i = 0; i < 8; ++i) a[8 + i] = (_Float16)p[8 + i];
  return a;
}

// 16-bit B 32x16: lanes 0-15 col N=lane hold K=0..15, lanes 16-31 hold K=16..31.
// W is (NOUT x K) row-major; B[k][n] = W[n][k].
// f16 pre-converted weights: one 32B-aligned v16h load, no converts.
__device__ __forceinline__ v16h ld_b_h16(const _Float16* W, int ldw, int n0, int k0, int lane) {
  int col = n0 + (lane & 15);
  int kh = ((lane >> 4) & 1) << 4;
  return *(const v16h*)(W + (long)col * ldw + k0 + kh);
}

__device__ __forceinline__ v16h ld_b_w32(const float* W, int ldw, int n0, int k0, int lane) {
  int col = n0 + (lane & 15);
  int kh = ((lane >> 4) & 1) << 4;
  const float* p = W + (long)col * ldw + k0 + kh;
  v16h b;
#pragma unroll
  for (int i = 0; i < 16; ++i) b[i] = (_Float16)p[i];
  return b;
}

__device__ __forceinline__ v8f gemm_ldsA_h16(const _Float16* A, int ldA, const _Float16* W,
                                             int ldw, int n0, int K, int lane) {
  v8f acc = {};
  for (int k0 = 0; k0 < K; k0 += 32) {
    v16h a = ld_a_lds(A, ldA, k0, lane);
    v16h b = ld_b_h16(W, ldw, n0, k0, lane);
    acc = __builtin_amdgcn_wmma_f32_16x16x32_f16(false, a, false, b, (short)0, acc, false, false);
  }
  return acc;
}

__device__ __forceinline__ v8f gemm_gA(const float* A, int ldA, int row0, const float* W, int ldw,
                                       int n0, int K, int lane) {
  v8f acc = {};
  for (int k0 = 0; k0 < K; k0 += 32) {
    v16h a = ld_a_g32(A, ldA, row0, k0, lane);
    v16h b = ld_b_w32(W, ldw, n0, k0, lane);
    acc = __builtin_amdgcn_wmma_f32_16x16x32_f16(false, a, false, b, (short)0, acc, false, false);
  }
  return acc;
}

// ---------------- small helper kernels ----------------
__global__ void k_zero(float* p, int n) {
  int i = blockIdx.x * 256 + threadIdx.x;
  if (i < n) p[i] = 0.0f;
}

__global__ void k_cvt(const float* __restrict__ src, _Float16* __restrict__ dst, int n) {
  int i = blockIdx.x * 256 + threadIdx.x;
  if (i < n) dst[i] = (_Float16)src[i];
}

__global__ void k_scatter(float* beta, const int* __restrict__ dst, int n) {
  int i = blockIdx.x * 256 + threadIdx.x;
  if (i < n) atomicAdd(&beta[dst[i]], 1.0f);
}

__global__ void k_hT(const float* __restrict__ h, float* __restrict__ hT) {
  int c = threadIdx.x;
  if (c < EMB) {
    float s = 0.0f;
    for (int t = 0; t < TT; ++t) s += h[t * EMB + c];
    hT[c] = s * (1.0f / TT);
  }
}

// ---------------- candidate scoring (dominant GEMM, WMMA f16) ----------------
__global__ __launch_bounds__(128) void k_score(
    const float* __restrict__ x, const float* __restrict__ h,
    const float* __restrict__ degree, const float* __restrict__ beta,
    const float* __restrict__ hT,
    const _Float16* __restrict__ rw, const float* __restrict__ rb,
    const float* __restrict__ nw, const float* __restrict__ nb_,
    const _Float16* __restrict__ m1w, const float* __restrict__ m1b,
    const float* __restrict__ m2w, const float* __restrict__ m2b,
    float* __restrict__ scores) {
  __shared__ _Float16 sX[16 * FEAT];   // 16KB  staged x tile
  __shared__ _Float16 sE[16 * 1024];   // 32KB  relu(concat) activations
  __shared__ _Float16 sH[16 * EMB];    // 8KB   relu hidden
  int tid = threadIdx.x, lane = tid & 31, wv = tid >> 5;
  int base = TT + blockIdx.x * 16;     // (N-T) = 16*6249 exactly

  for (int i = tid; i < 16 * FEAT; i += 128) {
    int r = i >> 9, c = i & 511;
    sX[i] = (_Float16)x[(long)(base + r) * FEAT + c];
  }
  for (int i = tid; i < 16 * 768; i += 128) {
    int r = i / 768, c = i - r * 768;
    int node = base + r;
    float v;
    if (c < 256)      v = h[(long)node * EMB + c];
    else if (c < 512) v = hT[c - 256];
    else {
      int n2 = c - 512;
      v = degree[node] * nw[n2 * 2 + 0] + beta[node] * nw[n2 * 2 + 1] + nb_[n2];
    }
    sE[r * 1024 + 256 + c] = (_Float16)fmaxf(v, 0.0f);
  }
  __syncthreads();

#pragma unroll
  for (int t = 0; t < 4; ++t) {        // GEMM1: x_v = x @ cs_raw_w^T, K=512
    int n0 = (wv * 4 + t) * 16;
    v8f acc = gemm_ldsA_h16(sX, FEAT, rw, FEAT, n0, FEAT, lane);
    int col = n0 + (lane & 15), mb = (lane >> 4) * 8;
#pragma unroll
    for (int r = 0; r < 8; ++r)
      sE[(mb + r) * 1024 + col] = (_Float16)fmaxf(acc[r] + rb[col], 0.0f);
  }
  __syncthreads();

#pragma unroll
  for (int t = 0; t < 4; ++t) {        // GEMM2: hidden = relu(e @ cs_m1_w^T), K=1024
    int n0 = (wv * 4 + t) * 16;
    v8f acc = gemm_ldsA_h16(sE, 1024, m1w, 1024, n0, 1024, lane);
    int col = n0 + (lane & 15), mb = (lane >> 4) * 8;
#pragma unroll
    for (int r = 0; r < 8; ++r)
      sH[(mb + r) * EMB + col] = (_Float16)fmaxf(acc[r] + m1b[col], 0.0f);
  }
  __syncthreads();

  if (tid < 16) {
    float s = m2b[0];
    for (int c = 0; c < EMB; ++c) s += (float)sH[tid * EMB + c] * m2w[c];
    scores[base - TT + tid] = s;
  }
}

// ---------------- top-128 (iterative argmax, low-index tie-break) ----------------
__global__ __launch_bounds__(1024) void k_topk(float* __restrict__ scores,
                                               int* __restrict__ cand, int M) {
  __shared__ float sv[1024];
  __shared__ int   si[1024];
  int tid = threadIdx.x;
  for (int r = 0; r < MAXC; ++r) {
    float bv = -1e30f; int bi = 0x7fffffff;
    for (int i = tid; i < M; i += 1024) {
      float v = scores[i];
      if (v > bv || (v == bv && i < bi)) { bv = v; bi = i; }
    }
    sv[tid] = bv; si[tid] = bi;
    __syncthreads();
    for (int s = 512; s > 0; s >>= 1) {
      if (tid < s) {
        float ov = sv[tid + s]; int oi = si[tid + s];
        if (ov > sv[tid] || (ov == sv[tid] && oi < si[tid])) { sv[tid] = ov; si[tid] = oi; }
      }
      __syncthreads();
    }
    if (tid == 0) {
      cand[r] = TT + si[0];
      scores[si[0]] = -1e30f;
    }
    __syncthreads();
  }
}

// ---------------- target / candidate encoders (WMMA, f16 weights) ----------------
__global__ __launch_bounds__(128) void k_encode(
    const float* __restrict__ x, const float* __restrict__ h, const float* __restrict__ degree,
    const int* __restrict__ idx_targets, const int* __restrict__ cand,
    const _Float16* te_rw, const float* te_rb, const float* te_nw, const float* te_nb,
    const _Float16* te_ow, const float* te_ob,
    const _Float16* ce_rw, const float* ce_rb, const float* ce_nw, const float* ce_nb,
    const _Float16* ce_ow, const float* ce_ob,
    const float* __restrict__ pos_emb, float* __restrict__ z) {
  __shared__ _Float16 sX[16 * FEAT];   // 16KB
  __shared__ _Float16 sE[16 * 768];    // 24KB
  __shared__ int ridx[16];
  int tid = threadIdx.x, lane = tid & 31, wv = tid >> 5;
  int blk = blockIdx.x;
  bool is_t = (blk == 0);
  const _Float16* rw = is_t ? te_rw : ce_rw;
  const float* rb = is_t ? te_rb : ce_rb;
  const float* nw = is_t ? te_nw : ce_nw;
  const float* nbv = is_t ? te_nb : ce_nb;
  const _Float16* ow = is_t ? te_ow : ce_ow;
  const float* ob = is_t ? te_ob : ce_ob;
  const float* pe = pos_emb + (is_t ? 0 : EMB);
  int rowbase = is_t ? 0 : TT + (blk - 1) * 16;

  if (tid < 16) ridx[tid] = is_t ? idx_targets[tid] : cand[(blk - 1) * 16 + tid];
  __syncthreads();

  for (int i = tid; i < 16 * FEAT; i += 128) {
    int r = i >> 9, c = i & 511;
    sX[i] = (_Float16)x[(long)ridx[r] * FEAT + c];
  }
  for (int i = tid; i < 16 * 512; i += 128) {
    int r = i >> 9, c = i & 511;
    int node = ridx[r];
    float v;
    if (c < 256) v = h[(long)node * EMB + c];
    else { int n2 = c - 256; v = degree[node] * nw[n2] + nbv[n2]; }
    sE[r * 768 + 256 + c] = (_Float16)fmaxf(v, 0.0f);
  }
  __syncthreads();

#pragma unroll
  for (int t = 0; t < 4; ++t) {        // xr = x @ rw^T, K=512
    int n0 = (wv * 4 + t) * 16;
    v8f acc = gemm_ldsA_h16(sX, FEAT, rw, FEAT, n0, FEAT, lane);
    int col = n0 + (lane & 15), mb = (lane >> 4) * 8;
#pragma unroll
    for (int r = 0; r < 8; ++r)
      sE[(mb + r) * 768 + col] = (_Float16)fmaxf(acc[r] + rb[col], 0.0f);
  }
  __syncthreads();

#pragma unroll
  for (int t = 0; t < 4; ++t) {        // out = enc @ ow^T, K=768
    int n0 = (wv * 4 + t) * 16;
    v8f acc = gemm_ldsA_h16(sE, 768, ow, 768, n0, 768, lane);
    int col = n0 + (lane & 15), mb = (lane >> 4) * 8;
#pragma unroll
    for (int r = 0; r < 8; ++r)
      z[(long)(rowbase + mb + r) * EMB + col] = acc[r] + ob[col] + pe[col];
  }
}

__global__ void k_zu(float* __restrict__ z, const float* __restrict__ z_u,
                     const float* __restrict__ pos_emb) {
  int tid = threadIdx.x;
  for (int i = tid; i < NBD * EMB; i += 256)
    z[(long)(START_ATK + i / EMB) * EMB + (i % EMB)] = z_u[i] + pos_emb[2 * EMB + (i % EMB)];
  for (int i = tid; i < (SEQP - SEQ) * EMB; i += 256)
    z[(long)SEQ * EMB + i] = 0.0f;     // zero pad rows used as WMMA A-rows
}

// ---------------- persistent single-WG transformer (6 layers, WMMA tiles) ----------------
__global__ __launch_bounds__(1024) void k_xformer(
    float* __restrict__ z, float* __restrict__ qkv, float* __restrict__ obuf,
    float* __restrict__ tbuf, float* __restrict__ fbuf,
    const float* __restrict__ qkv_w, const float* __restrict__ qkv_b,
    const float* __restrict__ ao_w, const float* __restrict__ ao_b,
    const float* __restrict__ ln1g, const float* __restrict__ ln1b,
    const float* __restrict__ ln2g, const float* __restrict__ ln2b,
    const float* __restrict__ f1w, const float* __restrict__ f1b,
    const float* __restrict__ f2w, const float* __restrict__ f2b) {
  int tid = threadIdx.x, lane = tid & 31, wv = tid >> 5;
  const int NW = 32;                    // 1024 threads = 32 wave32 waves
  for (int l = 0; l < LAY; ++l) {
    const float* Wq = qkv_w + (long)l * 768 * EMB;
    const float* Bq = qkv_b + (long)l * 768;
    for (int job = wv; job < 10 * 48; job += NW) {   // qkv: (149x256)@(256x768)
      int mt = job / 48, nt = job - mt * 48;
      v8f acc = gemm_gA(z, EMB, mt * 16, Wq, EMB, nt * 16, EMB, lane);
      int col = nt * 16 + (lane & 15), mb = mt * 16 + (lane >> 4) * 8;
#pragma unroll
      for (int r = 0; r < 8; ++r) {
        int m = mb + r;
        if (m < SEQ) qkv[(long)m * 768 + col] = acc[r] + Bq[col];
      }
    }
    __syncthreads();

    for (int p = tid; p < NH * SEQ; p += 1024) {     // attention (scalar, small)
      int hh = p / SEQ, s = p - hh * SEQ;
      const float scale = 0.125f;                    // 1/sqrt(64)
      const float* q = qkv + (long)s * 768 + hh * DH;
      float mx = -1e30f;
      for (int t2 = 0; t2 < SEQ; ++t2) {
        const float* kk = qkv + (long)t2 * 768 + 256 + hh * DH;
        float d = 0.0f;
#pragma unroll
        for (int c = 0; c < DH; ++c) d += q[c] * kk[c];
        mx = fmaxf(mx, d * scale);
      }
      float sum = 0.0f, o[DH];
#pragma unroll
      for (int c = 0; c < DH; ++c) o[c] = 0.0f;
      for (int t2 = 0; t2 < SEQ; ++t2) {
        const float* kk = qkv + (long)t2 * 768 + 256 + hh * DH;
        float d = 0.0f;
#pragma unroll
        for (int c = 0; c < DH; ++c) d += q[c] * kk[c];
        float w = __expf(d * scale - mx);
        sum += w;
        const float* vv = qkv + (long)t2 * 768 + 512 + hh * DH;
#pragma unroll
        for (int c = 0; c < DH; ++c) o[c] += w * vv[c];
      }
      float inv = 1.0f / sum;
#pragma unroll
      for (int c = 0; c < DH; ++c) obuf[(long)s * EMB + hh * DH + c] = o[c] * inv;
    }
    __syncthreads();

    const float* Wo = ao_w + (long)l * EMB * EMB;
    const float* Bo = ao_b + (long)l * EMB;
    for (int job = wv; job < 10 * 16; job += NW) {   // proj + residual
      int mt = job / 16, nt = job - mt * 16;
      v8f acc = gemm_gA(obuf, EMB, mt * 16, Wo, EMB, nt * 16, EMB, lane);
      int col = nt * 16 + (lane & 15), mb = mt * 16 + (lane >> 4) * 8;
#pragma unroll
      for (int r = 0; r < 8; ++r) {
        int m = mb + r;
        if (m < SEQ) tbuf[(long)m * EMB + col] = acc[r] + Bo[col] + z[(long)m * EMB + col];
      }
    }
    __syncthreads();

    if (tid < SEQ) {                                 // LN1 -> z
      const float* row = tbuf + (long)tid * EMB;
      float mean = 0.0f;
      for (int c = 0; c < EMB; ++c) mean += row[c];
      mean *= (1.0f / EMB);
      float var = 0.0f;
      for (int c = 0; c < EMB; ++c) { float d = row[c] - mean; var += d * d; }
      float inv = rsqrtf(var * (1.0f / EMB) + 1e-5f);
      for (int c = 0; c < EMB; ++c)
        z[(long)tid * EMB + c] = (row[c] - mean) * inv * ln1g[l * EMB + c] + ln1b[l * EMB + c];
    }
    __syncthreads();

    const float* W1 = f1w + (long)l * FFD * EMB;
    const float* B1 = f1b + (long)l * FFD;
    for (int job = wv; job < 10 * 64; job += NW) {   // ff1 + relu
      int mt = job / 64, nt = job - mt * 64;
      v8f acc = gemm_gA(z, EMB, mt * 16, W1, EMB, nt * 16, EMB, lane);
      int col = nt * 16 + (lane & 15), mb = mt * 16 + (lane >> 4) * 8;
#pragma unroll
      for (int r = 0; r < 8; ++r) {
        int m = mb + r;
        if (m < SEQ) fbuf[(long)m * FFD + col] = fmaxf(acc[r] + B1[col], 0.0f);
      }
    }
    __syncthreads();

    const float* W2 = f2w + (long)l * EMB * FFD;
    const float* B2 = f2b + (long)l * EMB;
    for (int job = wv; job < 10 * 16; job += NW) {   // ff2 + residual, K=1024
      int mt = job / 16, nt = job - mt * 16;
      v8f acc = gemm_gA(fbuf, FFD, mt * 16, W2, FFD, nt * 16, FFD, lane);
      int col = nt * 16 + (lane & 15), mb = mt * 16 + (lane >> 4) * 8;
#pragma unroll
      for (int r = 0; r < 8; ++r) {
        int m = mb + r;
        if (m < SEQ) tbuf[(long)m * EMB + col] = acc[r] + B2[col] + z[(long)m * EMB + col];
      }
    }
    __syncthreads();

    if (tid < SEQ) {                                 // LN2 -> z
      const float* row = tbuf + (long)tid * EMB;
      float mean = 0.0f;
      for (int c = 0; c < EMB; ++c) mean += row[c];
      mean *= (1.0f / EMB);
      float var = 0.0f;
      for (int c = 0; c < EMB; ++c) { float d = row[c] - mean; var += d * d; }
      float inv = rsqrtf(var * (1.0f / EMB) + 1e-5f);
      for (int c = 0; c < EMB; ++c)
        z[(long)tid * EMB + c] = (row[c] - mean) * inv * ln2g[l * EMB + c] + ln2b[l * EMB + c];
    }
    __syncthreads();
  }
}

// ---------------- final: fnorm + prelu, feature head, edge top-50 ----------------
__global__ __launch_bounds__(1024) void k_final(
    const float* __restrict__ z, float* __restrict__ zf, float* __restrict__ ez,
    const float* __restrict__ fg, const float* __restrict__ fb,
    const float* __restrict__ prelu_a,
    const float* __restrict__ feat_w, const float* __restrict__ feat_b,
    const float* __restrict__ min_feat, const float* __restrict__ max_feat,
    const float* __restrict__ edge_w, const float* __restrict__ edge_b,
    float* __restrict__ out) {
  __shared__ float sflat[NBD * SEQ];
  int tid = threadIdx.x;
  float pa = prelu_a[0];

  if (tid < SEQ) {                                   // fnorm LN + PReLU
    const float* row = z + (long)tid * EMB;
    float mean = 0.0f;
    for (int c = 0; c < EMB; ++c) mean += row[c];
    mean *= (1.0f / EMB);
    float var = 0.0f;
    for (int c = 0; c < EMB; ++c) { float d = row[c] - mean; var += d * d; }
    float inv = rsqrtf(var * (1.0f / EMB) + 1e-5f);
    for (int c = 0; c < EMB; ++c) {
      float v = (row[c] - mean) * inv * fg[c] + fb[c];
      zf[(long)tid * EMB + c] = (v >= 0.0f) ? v : pa * v;
    }
  }
  __syncthreads();

  for (int j = tid; j < NBD * FEAT; j += 1024) {     // x_atk = sigmoid(...)·range + min
    int i = j >> 9, f2 = j & 511;
    const float* zr = zf + (long)(START_ATK + i) * EMB;
    const float* wr = feat_w + (long)f2 * EMB;
    float d = feat_b[f2];
    for (int c = 0; c < EMB; ++c) d += zr[c] * wr[c];
    float sg = 1.0f / (1.0f + __expf(-d));
    out[i * OUTW + f2] = sg * (max_feat[f2] - min_feat[f2]) + min_feat[f2];
  }
  for (int j = tid; j < SEQ * EMB; j += 1024) {      // ez = z @ edge_w^T + b
    int m = j >> 8, n2 = j & 255;
    const float* zr = zf + (long)m * EMB;
    const float* wr = edge_w + (long)n2 * EMB;
    float d = edge_b[n2];
    for (int c = 0; c < EMB; ++c) d += zr[c] * wr[c];
    ez[m * EMB + n2] = d;
  }
  __syncthreads();

  for (int j = tid; j < NBD * SEQ; j += 1024) {      // masked scores
    int i = j / SEQ, s = j - i * SEQ;
    const float* zr = zf + (long)(START_ATK + i) * EMB;
    const float* er = ez + (long)s * EMB;
    float d = 0.0f;
    for (int c = 0; c < EMB; ++c) d += zr[c] * er[c];
    sflat[j] = (s <= START_ATK - 1 + i) ? d : -1e9f;
    out[i * OUTW + FEAT + s] = 0.0f;                 // init adj region
  }
  __syncthreads();

  if (tid == 0) {                                    // hard = top-50 mask
    for (int r = 0; r < EBD; ++r) {                  // (soft cancels in fwd pass)
      float bv = -2e9f; int bi = 0;
      for (int j = 0; j < NBD * SEQ; ++j)
        if (sflat[j] > bv) { bv = sflat[j]; bi = j; }
      sflat[bi] = -3e9f;
      int i = bi / SEQ, s = bi - i * SEQ;
      out[i * OUTW + FEAT + s] = 1.0f;
    }
  }
}

extern "C" void kernel_launch(void* const* d_in, const int* in_sizes, int n_in,
                              void* d_out, int out_size, void* d_ws, size_t ws_size,
                              hipStream_t stream) {
  const float* x        = (const float*)d_in[0];
  const float* h        = (const float*)d_in[1];
  const float* degree   = (const float*)d_in[2];
  const float* min_feat = (const float*)d_in[3];
  const float* max_feat = (const float*)d_in[4];
  const float* z_u      = (const float*)d_in[5];
  const int*   edges    = (const int*)d_in[6];
  const int*   idx_t    = (const int*)d_in[7];
  const float* cs_raw_w = (const float*)d_in[8];
  const float* cs_raw_b = (const float*)d_in[9];
  const float* cs_num_w = (const float*)d_in[10];
  const float* cs_num_b = (const float*)d_in[11];
  const float* cs_m1_w  = (const float*)d_in[12];
  const float* cs_m1_b  = (const float*)d_in[13];
  const float* cs_m2_w  = (const float*)d_in[14];
  const float* cs_m2_b  = (const float*)d_in[15];
  const float* te_raw_w = (const float*)d_in[16];
  const float* te_raw_b = (const float*)d_in[17];
  const float* te_num_w = (const float*)d_in[18];
  const float* te_num_b = (const float*)d_in[19];
  const float* te_out_w = (const float*)d_in[20];
  const float* te_out_b = (const float*)d_in[21];
  const float* ce_raw_w = (const float*)d_in[22];
  const float* ce_raw_b = (const float*)d_in[23];
  const float* ce_num_w = (const float*)d_in[24];
  const float* ce_num_b = (const float*)d_in[25];
  const float* ce_out_w = (const float*)d_in[26];
  const float* ce_out_b = (const float*)d_in[27];
  const float* pos_emb  = (const float*)d_in[28];
  const float* qkv_w    = (const float*)d_in[29];
  const float* qkv_b    = (const float*)d_in[30];
  const float* ao_w     = (const float*)d_in[31];
  const float* ao_b     = (const float*)d_in[32];
  const float* ln1g     = (const float*)d_in[33];
  const float* ln1b     = (const float*)d_in[34];
  const float* ln2g     = (const float*)d_in[35];
  const float* ln2b     = (const float*)d_in[36];
  const float* ff1_w    = (const float*)d_in[37];
  const float* ff1_b    = (const float*)d_in[38];
  const float* ff2_w    = (const float*)d_in[39];
  const float* ff2_b    = (const float*)d_in[40];
  const float* fnorm_g  = (const float*)d_in[41];
  const float* fnorm_b  = (const float*)d_in[42];
  const float* feat_w   = (const float*)d_in[43];
  const float* feat_b   = (const float*)d_in[44];
  const float* edge_w   = (const float*)d_in[45];
  const float* edge_b   = (const float*)d_in[46];
  const float* prelu_a  = (const float*)d_in[47];

  // ---------------- workspace carve ----------------
  // float region: 691,888 floats (2.77 MB) -- ends 32B-aligned
  float* ws     = (float*)d_ws;
  float* beta   = ws;                        // 100000
  float* hT     = beta + NN;                 // 256
  float* scores = hT + EMB;                  // 99984
  int*   cand   = (int*)(scores + (NN - TT));// 128
  float* z      = (float*)(cand + MAXC);     // 160*256
  float* qkv    = z + SEQP * EMB;            // 160*768
  float* obuf   = qkv + SEQP * 768;          // 160*256
  float* tbuf   = obuf + SEQP * EMB;         // 160*256
  float* fbuf   = tbuf + SEQP * EMB;         // 160*1024
  float* zf     = fbuf + SEQP * FFD;         // 160*256
  float* ez     = zf + SEQP * EMB;           // 160*256
  // f16 pre-converted weight region (2 MB), 32B-aligned
  _Float16* hw      = (_Float16*)(ez + SEQP * EMB);
  _Float16* h_csraw = hw;                          // 256*512
  _Float16* h_csm1  = h_csraw + EMB * FEAT;        // 256*1024
  _Float16* h_teraw = h_csm1  + EMB * FFD;         // 256*512
  _Float16* h_teout = h_teraw + EMB * FEAT;        // 256*768
  _Float16* h_ceraw = h_teout + EMB * 768;         // 256*512
  _Float16* h_ceout = h_ceraw + EMB * FEAT;        // 256*768

  // one-time (per launch) weight down-conversion to f16
  k_cvt<<<(EMB * FEAT + 255) / 256, 256, 0, stream>>>(cs_raw_w, h_csraw, EMB * FEAT);
  k_cvt<<<(EMB * FFD + 255) / 256, 256, 0, stream>>>(cs_m1_w, h_csm1, EMB * FFD);
  k_cvt<<<(EMB * FEAT + 255) / 256, 256, 0, stream>>>(te_raw_w, h_teraw, EMB * FEAT);
  k_cvt<<<(EMB * 768 + 255) / 256, 256, 0, stream>>>(te_out_w, h_teout, EMB * 768);
  k_cvt<<<(EMB * FEAT + 255) / 256, 256, 0, stream>>>(ce_raw_w, h_ceraw, EMB * FEAT);
  k_cvt<<<(EMB * 768 + 255) / 256, 256, 0, stream>>>(ce_out_w, h_ceout, EMB * 768);

  k_zero<<<(NN + 255) / 256, 256, 0, stream>>>(beta, NN);
  k_scatter<<<(ETGTN + 255) / 256, 256, 0, stream>>>(beta, edges, ETGTN);
  k_hT<<<1, 256, 0, stream>>>(h, hT);
  k_score<<<(NN - TT) / 16, 128, 0, stream>>>(x, h, degree, beta, hT,
      h_csraw, cs_raw_b, cs_num_w, cs_num_b, h_csm1, cs_m1_b, cs_m2_w, cs_m2_b, scores);
  k_topk<<<1, 1024, 0, stream>>>(scores, cand, NN - TT);
  k_encode<<<1 + MAXC / 16, 128, 0, stream>>>(x, h, degree, idx_t, cand,
      h_teraw, te_raw_b, te_num_w, te_num_b, h_teout, te_out_b,
      h_ceraw, ce_raw_b, ce_num_w, ce_num_b, h_ceout, ce_out_b, pos_emb, z);
  k_zu<<<1, 256, 0, stream>>>(z, z_u, pos_emb);
  k_xformer<<<1, 1024, 0, stream>>>(z, qkv, obuf, tbuf, fbuf,
      qkv_w, qkv_b, ao_w, ao_b, ln1g, ln1b, ln2g, ln2b, ff1_w, ff1_b, ff2_w, ff2_b);
  k_final<<<1, 1024, 0, stream>>>(z, zf, ez, fnorm_g, fnorm_b, prelu_a,
      feat_w, feat_b, min_feat, max_feat, edge_w, edge_b, (float*)d_out);
  (void)in_sizes; (void)n_in; (void)out_size; (void)ws_size;
}
